// RNN_56994216018397
// MI455X (gfx1250) — matrix operand order repaired and checked by reading, not compiled, not run on hardware
//
#include <hip/hip_runtime.h>
#include <math.h>

// Problem constants (match the reference setup)
#define SS 2048
#define BB 64
#define HH 256
#define LL 2

typedef float  v2f   __attribute__((ext_vector_type(2)));
typedef float  v8f   __attribute__((ext_vector_type(8)));
typedef __bf16 v8bf  __attribute__((ext_vector_type(8)));
typedef __bf16 v16bf __attribute__((ext_vector_type(16)));

__device__ __forceinline__ v8f wmma_f32(v2f a, v2f b, v8f c) {
    // V_WMMA_F32_16X16X4_F32 : exact f32 matrix op (recurrence path)
    return __builtin_amdgcn_wmma_f32_16x16x4_f32(
        false, a, false, b, (short)0, c, false, false);
}
__device__ __forceinline__ v8f wmma_bf16(v16bf a, v16bf b, v8f c) {
    // V_WMMA_F32_16X16X32_BF16 : 8x K-depth of the f32 op (xproj path)
    return __builtin_amdgcn_wmma_f32_16x16x32_bf16(
        false, a, false, b, (short)0, c, false, false);
}
__device__ __forceinline__ v16bf combine8(v8bf a, v8bf b) {
    return __builtin_shufflevector(a, b, 0, 1, 2, 3, 4, 5, 6, 7,
                                         8, 9, 10, 11, 12, 13, 14, 15);
}

// ---------------------------------------------------------------------------
// Prep: split f32 array into bf16 hi + lo (x == hi + lo to ~2^-16 rel).
// Done ONCE so phase 1 needs zero conversion VALU.
// ---------------------------------------------------------------------------
__global__ void __launch_bounds__(256) split_kernel(
    const float* __restrict__ src, __bf16* __restrict__ hi,
    __bf16* __restrict__ lo, int n)
{
    int i = blockIdx.x * 256 + threadIdx.x;
    if (i < n) {
        float x = src[i];
        __bf16 h = (__bf16)x;
        hi[i] = h;
        lo[i] = (__bf16)(x - (float)h);
    }
}

// ---------------------------------------------------------------------------
// Phase 1: x_proj[l,s,b,h] = sum_i seq[s,b,i] * Win[l,h,i]
// Split-precision bf16: x*y ~= hi*hi + hi*lo + lo*hi, f32 accumulate.
// One wave per 16x16 tile; K consumed 32 at a time.
// ---------------------------------------------------------------------------
__global__ void __launch_bounds__(256) xproj_kernel(
    const float*  __restrict__ seq,    // (S,B,H) f32
    const float*  __restrict__ Win,    // (L,H,H) f32 [n][k]
    const __bf16* __restrict__ seqHi,  // pre-split (or null -> convert inline)
    const __bf16* __restrict__ seqLo,
    const __bf16* __restrict__ WinHi,
    const __bf16* __restrict__ WinLo,
    float* __restrict__ xproj)         // (L,S,B,H)
{
    const int n_mtiles = (SS * BB) / 16;      // 8192
    const int n_ntiles = HH / 16;             // 16
    int wid  = blockIdx.x * 8 + (threadIdx.x >> 5);
    int lane = threadIdx.x & 31;

    int l  = wid / (n_mtiles * n_ntiles);
    int r  = wid % (n_mtiles * n_ntiles);
    int mt = r / n_ntiles;
    int nt = r % n_ntiles;
    if (l >= LL) return;

    const int half  = lane >> 4;              // 0: lanes 0-15, 1: lanes 16-31
    const int idx16 = lane & 15;
    const int m = mt * 16 + idx16;
    const int n = nt * 16 + idx16;

    v8f c = {};
    if (seqHi) {
        // A operand (16x32 bf16): lane holds K = c0+[0..7], c0+16+[0..7]; c0=8*half
        const __bf16* AH = seqHi + (size_t)m * HH + half * 8;
        const __bf16* AL = seqLo + (size_t)m * HH + half * 8;
        // B operand (32x16 bf16): lane holds K = c0+[0..7], c0+16+[0..7] of column n; c0=8*half
        const __bf16* BH = WinHi + (size_t)l * HH * HH + (size_t)n * HH + half * 8;
        const __bf16* BL = WinLo + (size_t)l * HH * HH + (size_t)n * HH + half * 8;
        for (int k0 = 0; k0 < HH; k0 += 32) {
            v16bf ah = combine8(*(const v8bf*)(AH + k0), *(const v8bf*)(AH + k0 + 16));
            v16bf al = combine8(*(const v8bf*)(AL + k0), *(const v8bf*)(AL + k0 + 16));
            v16bf bh = combine8(*(const v8bf*)(BH + k0), *(const v8bf*)(BH + k0 + 16));
            v16bf bl = combine8(*(const v8bf*)(BL + k0), *(const v8bf*)(BL + k0 + 16));
            c = wmma_bf16(ah, bh, c);
            c = wmma_bf16(ah, bl, c);
            c = wmma_bf16(al, bh, c);
        }
    } else {
        // fallback: convert f32 -> bf16 hi/lo on the fly
        const float* Arow = seq + (size_t)m * HH + half * 8;
        const float* Brow = Win + (size_t)l * HH * HH + (size_t)n * HH + half * 8;
        for (int k0 = 0; k0 < HH; k0 += 32) {
            v8f afa = *(const v8f*)(Arow + k0);
            v8f afb = *(const v8f*)(Arow + k0 + 16);
            v8f bfa = *(const v8f*)(Brow + k0);
            v8f bfb = *(const v8f*)(Brow + k0 + 16);
            v16bf ah, al, bh, bl;
#pragma unroll
            for (int i = 0; i < 8; ++i) {
                float x = afa[i]; __bf16 xh = (__bf16)x;
                ah[i]     = xh; al[i]     = (__bf16)(x - (float)xh);
                float y = afb[i]; __bf16 yh = (__bf16)y;
                ah[i + 8] = yh; al[i + 8] = (__bf16)(y - (float)yh);
                float u = bfa[i]; __bf16 uh = (__bf16)u;
                bh[i]     = uh; bl[i]     = (__bf16)(u - (float)uh);
                float w = bfb[i]; __bf16 wh = (__bf16)w;
                bh[i + 8] = wh; bl[i + 8] = (__bf16)(w - (float)wh);
            }
            c = wmma_bf16(ah, bh, c);
            c = wmma_bf16(ah, bl, c);
            c = wmma_bf16(al, bh, c);
        }
    }

    // C/D layout: element (m = vgpr + 8*half, n = lane%16)
    const int mo = half * 8;
    float* out = xproj + (size_t)l * SS * BB * HH + (size_t)(mt * 16) * HH + nt * 16 + idx16;
#pragma unroll
    for (int rr = 0; rr < 8; ++rr)
        out[(size_t)(mo + rr) * HH] = c[rr];
}

// ---------------------------------------------------------------------------
// Phase 2: sequential recurrence, one workgroup (32 waves) per layer.
// LDS (320KB = WGP max): Wt = Whid^T [k][n] 256KB, h_t = h^T [i][b] 64KB,
// bank-swizzled (col -> (col+row) & mask) to avoid 16-way conflicts.
// Each wave owns two M-tiles of one N-tile, sharing the B operand.
// ---------------------------------------------------------------------------
__device__ __forceinline__ int hidx(int i, int b) {          // h_t swizzle
    return i * BB + ((b + i) & (BB - 1));
}
__device__ __forceinline__ int widx(int k, int n) {          // Wt swizzle
    return k * HH + ((n + k) & (HH - 1));
}

__global__ void __launch_bounds__(1024) rnn_kernel(
    const float* __restrict__ seq,    // (S,B,H)
    const float* __restrict__ Win,    // (L,H,H)
    const float* __restrict__ Whid,   // (L,H,H)
    const float* __restrict__ init,   // (L,1,H)
    const float* __restrict__ xproj,  // (L,S,B,H) or nullptr -> fused path
    float* __restrict__ out_states,   // (S,B,H)   written by last layer
    float* __restrict__ out_finals)   // (L,B,H)
{
    extern __shared__ float smem[];
    float* Wt  = smem;            // HH*HH floats (256KB)
    float* h_t = smem + HH * HH;  // HH*BB floats (64KB)

    const int l    = blockIdx.x;
    const int tid  = threadIdx.x;
    const int wave = tid >> 5;
    const int lane = tid & 31;

    const float* Wh = Whid + (size_t)l * HH * HH;
    const float* Wi = Win  + (size_t)l * HH * HH;

    // Transpose Whid into LDS: coalesced global reads, swizzled writes
    for (int j = tid; j < HH * HH; j += 1024) {
        int nrow = j >> 8, kcol = j & (HH - 1);
        Wt[widx(kcol, nrow)] = Wh[j];
    }
    // h0 broadcast over batch, transposed
    for (int j = tid; j < HH * BB; j += 1024) {
        int i = j >> 6, b = j & (BB - 1);
        h_t[hidx(i, b)] = init[l * HH + i];
    }
    __syncthreads();

    const int half  = lane >> 4;
    const int idx16 = lane & 15;
    const int koff  = half * 2;           // f32 A/B: lanes 16-31 hold K=2,3
    const int mo    = half * 8;
    const bool last_layer = (l == LL - 1);

    const int nt  = wave & 15;            // both tiles share the N-tile
    const int mt0 = (wave >> 4) * 2, mt1 = mt0 + 1;
    const int m0  = mt0 * 16 + idx16, m1 = mt1 * 16 + idx16;
    const int n   = nt * 16 + idx16;

    const float* BW = Wi + (size_t)n * HH + koff;   // fused-path B operand (global)

    for (int s = 0; s < SS; ++s) {
        v8f c0 = {}, c1 = {};

        // hidden contribution: h @ Whid^T, all-LDS operands, B shared by both tiles
        for (int k0 = 0; k0 < HH; k0 += 4) {
            int k = k0 + koff;
            v2f b;  b[0]  = Wt[widx(k, n)];    b[1]  = Wt[widx(k + 1, n)];
            v2f a0; a0[0] = h_t[hidx(k, m0)];  a0[1] = h_t[hidx(k + 1, m0)];
            v2f a1; a1[0] = h_t[hidx(k, m1)];  a1[1] = h_t[hidx(k + 1, m1)];
            c0 = wmma_f32(a0, b, c0);
            c1 = wmma_f32(a1, b, c1);
        }

        if (xproj) {
            const float* xp = xproj + ((size_t)l * SS + s) * BB * HH;
            __builtin_prefetch(xp + BB * HH, 0, 0);   // next step -> global_prefetch_b8
#pragma unroll
            for (int rr = 0; rr < 8; ++rr) {
                c0[rr] += xp[(size_t)(mt0 * 16 + mo + rr) * HH + nt * 16 + idx16];
                c1[rr] += xp[(size_t)(mt1 * 16 + mo + rr) * HH + nt * 16 + idx16];
            }
        } else {
            // fused input projection: x[s] @ Win^T (exact f32 WMMA, L2-served)
            const float* X   = seq + (size_t)s * BB * HH;
            const float* XA0 = X + (size_t)m0 * HH + koff;
            const float* XA1 = X + (size_t)m1 * HH + koff;
            for (int k0 = 0; k0 < HH; k0 += 4) {
                v2f b  = *(const v2f*)(BW  + k0);
                v2f a0 = *(const v2f*)(XA0 + k0);
                v2f a1 = *(const v2f*)(XA1 + k0);
                c0 = wmma_f32(a0, b, c0);
                c1 = wmma_f32(a1, b, c1);
            }
        }

        // exact gelu: 0.5*x*(1+erf(x/sqrt(2)))
#pragma unroll
        for (int rr = 0; rr < 8; ++rr) {
            float x = c0[rr];
            c0[rr] = 0.5f * x * (1.0f + erff(x * 0.70710678118654752f));
            float y = c1[rr];
            c1[rr] = 0.5f * y * (1.0f + erff(y * 0.70710678118654752f));
        }

        __syncthreads();  // everyone done reading old h
#pragma unroll
        for (int rr = 0; rr < 8; ++rr) {
            // element (batch m, hidden ne) -> h_t[ne][m], swizzled
            h_t[hidx(nt * 16 + idx16, mt0 * 16 + mo + rr)] = c0[rr];
            h_t[hidx(nt * 16 + idx16, mt1 * 16 + mo + rr)] = c1[rr];
        }
        if (last_layer) {
            float* so = out_states + (size_t)s * BB * HH;
#pragma unroll
            for (int rr = 0; rr < 8; ++rr) {
                so[(size_t)(mt0 * 16 + mo + rr) * HH + nt * 16 + idx16] = c0[rr];
                so[(size_t)(mt1 * 16 + mo + rr) * HH + nt * 16 + idx16] = c1[rr];
            }
        }
        __syncthreads();  // h_new visible before next step
    }

    // final hidden for this layer (h_t holds h_last, transposed+swizzled)
    for (int j = tid; j < BB * HH; j += 1024) {
        int b = j >> 8, i = j & (HH - 1);
        out_finals[(size_t)l * BB * HH + j] = h_t[hidx(i, b)];
    }
}

extern "C" void kernel_launch(void* const* d_in, const int* in_sizes, int n_in,
                              void* d_out, int out_size, void* d_ws, size_t ws_size,
                              hipStream_t stream) {
    const float* seq  = (const float*)d_in[0];   // (S,B,H)
    const float* Win  = (const float*)d_in[1];   // (L,H,H)
    const float* Whid = (const float*)d_in[2];   // (L,H,H)
    const float* init = (const float*)d_in[3];   // (L,1,H)
    float* out = (float*)d_out;                  // states (S,B,H) ++ finals (L,B,H)

    const size_t nSeq  = (size_t)SS * BB * HH;        // 33.5M elems
    const size_t nWin  = (size_t)LL * HH * HH;        // 131072 elems
    const size_t xprojBytes = (size_t)LL * nSeq * sizeof(float);           // 268 MB
    const size_t splitBytes = 2 * nSeq * sizeof(__bf16)                    // 134 MB
                            + 2 * nWin * sizeof(__bf16);                   // + 0.5 MB

    float* xproj = nullptr;
    __bf16 *sHi = nullptr, *sLo = nullptr, *wHi = nullptr, *wLo = nullptr;

    if (ws_size >= xprojBytes) {
        xproj = (float*)d_ws;
        if (ws_size >= xprojBytes + splitBytes) {
            char* p = (char*)d_ws + xprojBytes;
            sHi = (__bf16*)p;                 p += nSeq * sizeof(__bf16);
            sLo = (__bf16*)p;                 p += nSeq * sizeof(__bf16);
            wHi = (__bf16*)p;                 p += nWin * sizeof(__bf16);
            wLo = (__bf16*)p;
            split_kernel<<<(int)((nSeq + 255) / 256), 256, 0, stream>>>(seq, sHi, sLo, (int)nSeq);
            split_kernel<<<(int)((nWin + 255) / 256), 256, 0, stream>>>(Win, wHi, wLo, (int)nWin);
        }
        const int n_waves  = LL * ((SS * BB) / 16) * (HH / 16);  // 262144
        const int n_blocks = n_waves / 8;                        // 32768
        xproj_kernel<<<n_blocks, 256, 0, stream>>>(seq, Win, sHi, sLo, wHi, wLo, xproj);
    }

    const int lds_bytes = (HH * HH + HH * BB) * (int)sizeof(float);  // 320KB (WGP max)
    (void)hipFuncSetAttribute((const void*)rnn_kernel,
                              hipFuncAttributeMaxDynamicSharedMemorySize, lds_bytes);
    rnn_kernel<<<LL, 1024, lds_bytes, stream>>>(seq, Win, Whid, init, xproj,
                                                out, out + (size_t)SS * BB * HH);
}
